// DecomposableAttentionModel_69982197121134
// MI455X (gfx1250) — matrix operand order repaired and checked by reading, not compiled, not run on hardware
//
#include <hip/hip_runtime.h>
#include <cstdint>
#include <cstddef>

typedef __attribute__((ext_vector_type(16))) _Float16 v16h;
typedef __attribute__((ext_vector_type(8)))  float    v8f;
typedef __attribute__((ext_vector_type(4)))  unsigned int v4u;
typedef __attribute__((ext_vector_type(4)))  int      v4i;
typedef __attribute__((ext_vector_type(8)))  int      v8i;

#define Bsz 8
#define Nn  128
#define NODE_DIM 64
#define Ll  512
#define RES_DIM 1024
#define CG 128
#define CP 128
#define HID 192   // CG + CP/2

#if defined(__HIP_DEVICE_COMPILE__) && __has_builtin(__builtin_amdgcn_tensor_load_to_lds)
#define USE_TDM 1
#else
#define USE_TDM 0
#endif

// ---------------------------------------------------------------- utilities
__global__ void f2h_kernel(const float* __restrict__ s, _Float16* __restrict__ d, int n) {
    int i = blockIdx.x * blockDim.x + threadIdx.x;
    if (i < n) d[i] = (_Float16)s[i];
}

// norm[b*N+n] = clip(deg,1)^-0.5
__global__ void norm_kernel(const float* __restrict__ adj, float* __restrict__ norm) {
    int idx = blockIdx.x * blockDim.x + threadIdx.x;   // b*N + n
    if (idx < Bsz * Nn) {
        const float* row = adj + (size_t)idx * Nn;
        float d = 0.f;
        for (int j = 0; j < Nn; ++j) d += row[j];
        if (d < 1.f) d = 1.f;
        norm[idx] = rsqrtf(d);
    }
}

// hs16 = f16(h * norm_row)   (h flat [B*N, CG])
__global__ void scale_h16_kernel(const float* __restrict__ h, const float* __restrict__ norm,
                                 _Float16* __restrict__ hs, int total) {
    int i = blockIdx.x * blockDim.x + threadIdx.x;
    if (i < total) hs[i] = (_Float16)(h[i] * norm[i >> 7]);   // CG=128 per row
}

#if USE_TDM
// Issue a 2D TDM tile load: rows x width elements into LDS (compact row-major).
// dsz: 1=2B, 2=4B element codes per D# spec.
__device__ __forceinline__ void tdm_load_2d(unsigned lds_off,
                                            unsigned long long ga,
                                            unsigned width, unsigned rows,
                                            unsigned stride, unsigned total_rows,
                                            unsigned dsz) {
    v4u g0 = { 1u,                                          // count=1, no gather
               lds_off,
               (unsigned)(ga & 0xFFFFFFFFull),
               (unsigned)((ga >> 32) & 0x01FFFFFFull) | 0x80000000u };   // type=2
    v8i g1 = { (int)(dsz << 16),                            // data_size
               (int)((stride & 0xFFFFu) << 16),             // tensor_dim0 lo16
               (int)((stride >> 16) | ((total_rows & 0xFFFFu) << 16)), // d0 hi|d1 lo
               (int)((total_rows >> 16) | (width << 16)),   // d1 hi | tile_dim0
               (int)rows,                                   // tile_dim1, tile_dim2=0
               (int)stride,                                 // tensor_dim0_stride lo32
               0, 0 };
    v4i z4 = { 0, 0, 0, 0 };
#if __clang_major__ >= 23
    v8i z8 = { 0, 0, 0, 0, 0, 0, 0, 0 };
    __builtin_amdgcn_tensor_load_to_lds(g0, g1, z4, z4, z8, 0);
#else
    __builtin_amdgcn_tensor_load_to_lds(g0, g1, z4, z4, 0);
#endif
}
#endif

// --------------------------------------------------- tiled WMMA f16 GEMM
// D[M,N] = act( rowscale(A[M,K]) * B16[K,N] + bias )
// Block = 128 threads (4 waves), 64x64 output tile, K chunked by 32.
// A tile: coalesced float4 loads + f16 convert, row-major in LDS.
// B tile: coalesced float4 loads, stored TRANSPOSED (BsT[col][k]) so both
// operand fragments are two 8-half runs -> ds_load_b128 pairs.
__global__ void gemm_wmma_f16(const float* __restrict__ A, long long sAb, int lda,
                              const _Float16* __restrict__ Bm, long long sBb, int ldb,
                              const float* __restrict__ rowscale, long long sRb,
                              const float* __restrict__ bias,
                              float* __restrict__ D, long long sDb, int ldd,
                              int K, int act)
{
    __shared__ __align__(16) _Float16 As[64][32];    // 4 KB, row-major
    __shared__ __align__(16) _Float16 BsT[64][32];   // 4 KB, col-major (BsT[n][k])

    const int batch = blockIdx.z;
    A  += (size_t)batch * sAb;
    Bm += (size_t)batch * sBb;
    D  += (size_t)batch * sDb;
    const float* rs = rowscale ? rowscale + (size_t)batch * sRb : nullptr;

    const int row0 = blockIdx.x * 64;
    const int n0   = blockIdx.y * 64;
    const int tid  = threadIdx.x;
    const int w    = tid >> 5;
    const int lane = tid & 31;
    const int mn   = lane & 15;
    const int hi   = lane >> 4;
    const int kbA  = hi << 3;        // A: K groups {0-7,16-23} / {8-15,24-31}
    const int kbB  = hi << 4;        // unused: B fragment is read with kbA, same k groups as A

    v8f acc[4] = {};

    for (int k0 = 0; k0 < K; k0 += 32) {
        __syncthreads();   // previous-iteration readers done with As/BsT

        // ---- stage A tile (64x32 f32 -> f16, rowscaled), 512 float4 chunks
        for (int idx = tid; idx < 512; idx += 128) {
            const int rr = idx >> 3;
            const int c4 = (idx & 7) << 2;
            const float* src = A + (size_t)(row0 + rr) * lda + k0 + c4;
            const float s = rs ? rs[row0 + rr] : 1.0f;
            float4 vlv = *(const float4*)src;
            As[rr][c4 + 0] = (_Float16)(vlv.x * s);
            As[rr][c4 + 1] = (_Float16)(vlv.y * s);
            As[rr][c4 + 2] = (_Float16)(vlv.z * s);
            As[rr][c4 + 3] = (_Float16)(vlv.w * s);
            if (k0 + 32 < K) __builtin_prefetch(src + 32, 0, 1);   // next k-tile
        }

        // ---- stage B tile (32x64 f16), coalesced reads, transposed stores
        for (int idx = tid; idx < 256; idx += 128) {
            const int rr = idx >> 3;          // K row 0..31
            const int c8 = (idx & 7) << 3;    // col 0..56
            float4 q = *(const float4*)(Bm + (size_t)(k0 + rr) * ldb + n0 + c8);
            const _Float16* hh = (const _Float16*)&q;
#pragma unroll
            for (int j = 0; j < 8; ++j) BsT[c8 + j][rr] = hh[j];
        }
        __syncthreads();

        // ---- fragments from LDS, 4 WMMAs per wave sharing one A fragment
        v16h a = {};
#pragma unroll
        for (int i = 0; i < 8; ++i) {
            a[i]     = As[w * 16 + mn][kbA + i];
            a[i + 8] = As[w * 16 + mn][kbA + 16 + i];
        }
#pragma unroll
        for (int nt = 0; nt < 4; ++nt) {
            v16h b = {};
#pragma unroll
            for (int i = 0; i < 8; ++i) {
                b[i]     = BsT[nt * 16 + mn][kbA + i];
                b[i + 8] = BsT[nt * 16 + mn][kbA + 16 + i];
            }
            acc[nt] = __builtin_amdgcn_wmma_f32_16x16x32_f16(
                false, a, false, b, (short)0, acc[nt], false, false);
        }
    }

    // ---- epilogue: bias + activation, D layout per f32 C/D VGPR table
#pragma unroll
    for (int nt = 0; nt < 4; ++nt) {
        const int col = n0 + nt * 16 + mn;
        const float bv = bias ? bias[col] : 0.0f;
#pragma unroll
        for (int v = 0; v < 8; ++v) {
            const int mrow = row0 + w * 16 + ((lane < 16) ? v : v + 8);
            float val = acc[nt][v] + bv;
            if (act == 1)      val = val > 0.f ? val : 0.f;
            else if (act == 2) val = tanhf(val);
            D[(size_t)mrow * ldd + col] = val;
        }
    }
}

// ------------------------------------------------- pair relu-sum reduction
// P[b, chunk, c] = sum_{n in chunk(16), l} relu(u[b,n,c] + v[b,l,c])
// v streamed through double-buffered LDS tiles by the Tensor Data Mover,
// pipelined one tile ahead of the VALU reduction.
__global__ void pairsum_kernel(const float* __restrict__ u, const float* __restrict__ v,
                               float* __restrict__ P)
{
    __shared__ float vs[2][32][HID];          // 2 x 24 KB tiles
    const int b = blockIdx.x;
    const int chunk = blockIdx.y;
    const int c = threadIdx.x;                // 0..191 (one channel)
    const float* vsrc = v + (size_t)b * Ll * HID;

    float uc[16];
#pragma unroll
    for (int nn = 0; nn < 16; ++nn)
        uc[nn] = u[((size_t)b * Nn + chunk * 16 + nn) * HID + c];

    float acc = 0.f;
#if USE_TDM
    if (threadIdx.x < 32)
        tdm_load_2d((unsigned)(uintptr_t)&vs[0][0][0],
                    (unsigned long long)(uintptr_t)vsrc,
                    HID, 32, HID, Ll, 2u);
#endif
    for (int t = 0; t < Ll / 32; ++t) {
#if USE_TDM
        if (threadIdx.x < 32) {
            if (t + 1 < Ll / 32) {
                tdm_load_2d((unsigned)(uintptr_t)&vs[(t + 1) & 1][0][0],
                            (unsigned long long)(uintptr_t)(vsrc + (size_t)(t + 1) * 32 * HID),
                            HID, 32, HID, Ll, 2u);
                __builtin_amdgcn_s_wait_tensorcnt(1);   // tile t resident (in-order)
            } else {
                __builtin_amdgcn_s_wait_tensorcnt(0);
            }
        }
#else
        for (int i = threadIdx.x; i < 32 * HID; i += blockDim.x)
            vs[t & 1][i / HID][i % HID] = vsrc[(size_t)(t * 32 + i / HID) * HID + i % HID];
#endif
        __syncthreads();
        const float* vb = &vs[t & 1][0][0];
#pragma unroll 4
        for (int l = 0; l < 32; ++l) {
            const float vv = vb[l * HID + c];
#pragma unroll
            for (int nn = 0; nn < 16; ++nn) {
                float x = uc[nn] + vv;
                acc += (x > 0.f) ? x : 0.f;
            }
        }
        __syncthreads();
    }
    P[((size_t)b * 8 + chunk) * HID + c] = acc;
}

// ----------------------------------------------------------------- finalize
// out[b] = S[b,:]·(W2@Wo) + N*L*(b2·Wo) + bo
__global__ void finalize_kernel(const float* __restrict__ P,
                                const float* __restrict__ W2, const float* __restrict__ b2,
                                const float* __restrict__ Wo, const float* __restrict__ bo,
                                float* __restrict__ out)
{
    __shared__ float w2o[HID];
    __shared__ float red[256];
    const int t = threadIdx.x;                // 256 threads
    if (t < HID) {
        float s = 0.f;
        for (int c = 0; c < CG; ++c) s += W2[t * CG + c] * Wo[c];
        w2o[t] = s;
    }
    __syncthreads();
    for (int b = 0; b < Bsz; ++b) {
        float s = 0.f;
        if (t < HID) {
            float S = 0.f;
            for (int ch = 0; ch < 8; ++ch) S += P[((size_t)b * 8 + ch) * HID + t];
            s = S * w2o[t];
        }
        red[t] = s;
        __syncthreads();
        for (int off = 128; off > 0; off >>= 1) {
            if (t < off) red[t] += red[t + off];
            __syncthreads();
        }
        if (t == 0) {
            float base = 0.f;
            for (int c = 0; c < CG; ++c) base += b2[c] * Wo[c];
            out[b] = red[0] + (float)(Nn * Ll) * base + bo[0];
        }
        __syncthreads();
    }
}

// ------------------------------------------------------------------- launch
static inline char* carve(char*& p, size_t bytes) {
    char* r = p;
    p += (bytes + 255) & ~(size_t)255;
    return r;
}

extern "C" void kernel_launch(void* const* d_in, const int* in_sizes, int n_in,
                              void* d_out, int out_size, void* d_ws, size_t ws_size,
                              hipStream_t stream) {
    const float* adj  = (const float*)d_in[0];   // [8,128,128]
    const float* nodes= (const float*)d_in[1];   // [8,128,64]
    const float* prot = (const float*)d_in[2];   // [8,512,1024]
    const float* Wn   = (const float*)d_in[3];   // [64,128]
    const float* bn   = (const float*)d_in[4];
    const float* Wg   = (const float*)d_in[5];   // [128,128]
    const float* bg   = (const float*)d_in[6];
    const float* Wr   = (const float*)d_in[7];   // [1024,128]
    const float* br   = (const float*)d_in[8];
    // d_in[9]=Wa, d_in[10]=ba: dead (softmax over size-1 axis == 1)
    const float* W1   = (const float*)d_in[11];  // [256,192]
    const float* b1   = (const float*)d_in[12];  // [192]
    const float* W2   = (const float*)d_in[13];  // [192,128]
    const float* b2   = (const float*)d_in[14];
    const float* Wo   = (const float*)d_in[15];  // [128,1]
    const float* bo   = (const float*)d_in[16];
    float* out = (float*)d_out;

    char* p = (char*)d_ws;
    float*    h    = (float*)   carve(p, (size_t)Bsz*Nn*CG     * 4);
    float*    m    = (float*)   carve(p, (size_t)Bsz*Nn*CG     * 4);
    float*    r    = (float*)   carve(p, (size_t)Bsz*Ll*CP     * 4);
    float*    u    = (float*)   carve(p, (size_t)Bsz*Nn*HID    * 4);
    float*    v    = (float*)   carve(p, (size_t)Bsz*Ll*HID    * 4);
    float*    norm = (float*)   carve(p, (size_t)Bsz*Nn        * 4);
    float*    P    = (float*)   carve(p, (size_t)Bsz*8*HID     * 4);
    _Float16* hs16 = (_Float16*)carve(p, (size_t)Bsz*Nn*CG     * 2);
    _Float16* Wn16 = (_Float16*)carve(p, (size_t)NODE_DIM*CG   * 2);
    _Float16* Wg16 = (_Float16*)carve(p, (size_t)CG*CG         * 2);
    _Float16* Wr16 = (_Float16*)carve(p, (size_t)RES_DIM*CP    * 2);
    _Float16* W116 = (_Float16*)carve(p, (size_t)(CG+CP)*HID   * 2);
    (void)ws_size; (void)n_in; (void)in_sizes; (void)out_size;

    auto cvt = [&](const float* s, _Float16* d, int n) {
        f2h_kernel<<<dim3((n + 255) / 256), dim3(256), 0, stream>>>(s, d, n);
    };
    cvt(Wn, Wn16, NODE_DIM * CG);
    cvt(Wg, Wg16, CG * CG);
    cvt(Wr, Wr16, RES_DIM * CP);
    cvt(W1, W116, (CG + CP) * HID);

    norm_kernel<<<dim3(4), dim3(256), 0, stream>>>(adj, norm);

    // h0 = nodes @ Wn + bn     [1024,64]x[64,128]
    gemm_wmma_f16<<<dim3(Bsz*Nn/64, CG/64, 1), dim3(128), 0, stream>>>(
        nodes, 0, NODE_DIM, Wn16, 0, CG, nullptr, 0, bn, h, 0, CG, NODE_DIM, 0);

    // 3 GNN steps: m = adj @ f16(h*norm);  h = act((m*norm) @ Wg + bg)
    for (int i = 0; i < 3; ++i) {
        scale_h16_kernel<<<dim3(Bsz*Nn*CG/256), dim3(256), 0, stream>>>(h, norm, hs16, Bsz*Nn*CG);
        gemm_wmma_f16<<<dim3(Nn/64, CG/64, Bsz), dim3(128), 0, stream>>>(
            adj, (long long)Nn*Nn, Nn, hs16, (long long)Nn*CG, CG,
            nullptr, 0, nullptr, m, (long long)Nn*CG, CG, Nn, 0);
        gemm_wmma_f16<<<dim3(Nn/64, CG/64, Bsz), dim3(128), 0, stream>>>(
            m, (long long)Nn*CG, CG, Wg16, 0, CG,
            norm, Nn, bg, h, (long long)Nn*CG, CG, CG, (i < 2) ? 1 : 2);
    }

    // r = prot @ Wr + br       [4096,1024]x[1024,128]
    gemm_wmma_f16<<<dim3(Bsz*Ll/64, CP/64, 1), dim3(128), 0, stream>>>(
        prot, 0, RES_DIM, Wr16, 0, CP, nullptr, 0, br, r, 0, CP, RES_DIM, 0);

    // u = h @ W1[:128]         [1024,128]x[128,192]   (no bias)
    gemm_wmma_f16<<<dim3(Bsz*Nn/64, HID/64, 1), dim3(128), 0, stream>>>(
        h, 0, CG, W116, 0, HID, nullptr, 0, nullptr, u, 0, HID, CG, 0);

    // v = r @ W1[128:] + b1    [4096,128]x[128,192]
    gemm_wmma_f16<<<dim3(Bsz*Ll/64, HID/64, 1), dim3(128), 0, stream>>>(
        r, 0, CP, W116 + (size_t)CG * HID, 0, HID, nullptr, 0, b1, v, 0, HID, CP, 0);

    // P[b,chunk,c] = sum over (16 nodes)x(512 residues) of relu(u+v)
    pairsum_kernel<<<dim3(Bsz, 8), dim3(HID), 0, stream>>>(u, v, P);

    // out[b] = S·(W2@Wo) + N*L*(b2·Wo) + bo
    finalize_kernel<<<dim3(1), dim3(256), 0, stream>>>(P, W2, b2, Wo, bo, out);
}
